// GraphEncoder_19774029431188
// MI455X (gfx1250) — compile-verified, matrix operand order checked by reading, not run on hardware
//
#include <hip/hip_runtime.h>

// ---------------------------------------------------------------------------
// Problem constants (from the reference)
// ---------------------------------------------------------------------------
constexpr int N_ = 20000;
constexpr int E_ = 320000;
constexpr int D_ = 128;
constexpr int H_ = 8;
constexpr int HD_ = 16;

typedef __bf16 bf16_t;
typedef __attribute__((ext_vector_type(16))) __bf16 v16bf;
typedef __attribute__((ext_vector_type(8)))  float  v8f;

// ---------------------------------------------------------------------------
// Device helpers
// ---------------------------------------------------------------------------
__device__ __forceinline__ bf16_t f2bf(float f) {
    unsigned u = __float_as_uint(f);
    unsigned r = (u + 0x7FFFu + ((u >> 16) & 1u)) >> 16;   // round-to-nearest-even
    unsigned short s = (unsigned short)r;
    bf16_t b;
    __builtin_memcpy(&b, &s, 2);
    return b;
}
__device__ __forceinline__ float bf2f(bf16_t b) {
    unsigned short s;
    __builtin_memcpy(&s, &b, 2);
    return __uint_as_float(((unsigned)s) << 16);
}
// order-preserving float<->uint for atomic max on signed floats
__device__ __forceinline__ unsigned f2ord(float f) {
    unsigned b = __float_as_uint(f);
    return (b & 0x80000000u) ? ~b : (b | 0x80000000u);
}
__device__ __forceinline__ float ord2f(unsigned k) {
    return (k & 0x80000000u) ? __uint_as_float(k ^ 0x80000000u)
                             : __uint_as_float(~k);
}
__device__ __forceinline__ float wave_sum(float v) {
    #pragma unroll
    for (int off = 16; off > 0; off >>= 1) v += __shfl_xor(v, off, 32);
    return v;
}

// ---------------------------------------------------------------------------
// bf16 WMMA GEMM:  C[M x 128] = A[M x KP] * Wp (col-packed) + bias
//
// Block = 256 threads (8 waves) computing a 32x128 output stripe:
//   - the 32xKP A stripe is staged in LDS once (cooperative b128 loads),
//   - each wave owns one 16-column tile and two 16-row tiles,
//   - B fragments are loaded once per wave and reused for both row tiles,
//   - KP is a compile-time constant so all fragment loads are hoisted ahead
//     of the back-to-back v_wmma chain (accumulate chaining is hazard-free).
// Grid.x = M/32 (M = 20000 or 320000, both divisible by 32).
// ---------------------------------------------------------------------------
template <int KP, bool OUT_BF16>
__global__ __launch_bounds__(256) void gemm_bf16_128(
    const bf16_t* __restrict__ A, const bf16_t* __restrict__ Wp,
    const float* __restrict__ bias, void* __restrict__ Cout)
{
    __shared__ bf16_t As[32 * KP];

    const int t    = threadIdx.x;
    const int row0 = blockIdx.x << 5;                 // 32 rows per block

    // ---- stage A stripe to LDS (16B chunks) ----
    constexpr int CHUNKS = 32 * KP / 8;               // # of 16-byte chunks
    #pragma unroll
    for (int c = t; c < CHUNKS; c += 256) {
        int rowl = c / (KP / 8);
        int off  = (c - rowl * (KP / 8)) * 8;
        *(uint4*)(As + rowl * KP + off) =
            *(const uint4*)(A + (size_t)(row0 + rowl) * KP + off);
    }
    __syncthreads();

    const int wid   = t >> 5;
    const int lane  = t & 31;
    const int col0  = wid << 4;
    const int l16   = lane & 15;
    const int khalf = (lane >> 4) << 3;               // 0 or 8

    constexpr int NK = KP / 32;
    union Frag { uint4 u[2]; v16bf v; };

    // ---- B fragments: load all K-steps up front (global, clause-grouped) ----
    const bf16_t* bcol = Wp + (size_t)(col0 + l16) * KP + khalf;
    Frag bfr[NK];
    #pragma unroll
    for (int s = 0; s < NK; ++s) {
        bfr[s].u[0] = *(const uint4*)(bcol + s * 32);
        bfr[s].u[1] = *(const uint4*)(bcol + s * 32 + 16);
    }

    // ---- two row tiles per wave, A fragments from LDS ----
    const bf16_t* arow0 = As + (size_t)l16 * KP + khalf;
    const bf16_t* arow1 = As + (size_t)(l16 + 16) * KP + khalf;

    v8f acc0 = {0.f, 0.f, 0.f, 0.f, 0.f, 0.f, 0.f, 0.f};
    v8f acc1 = {0.f, 0.f, 0.f, 0.f, 0.f, 0.f, 0.f, 0.f};
    #pragma unroll
    for (int s = 0; s < NK; ++s) {
        Frag a0;
        a0.u[0] = *(const uint4*)(arow0 + s * 32);
        a0.u[1] = *(const uint4*)(arow0 + s * 32 + 16);
        acc0 = __builtin_amdgcn_wmma_f32_16x16x32_bf16(
            false, a0.v, false, bfr[s].v, (short)0, acc0, false, false);
    }
    #pragma unroll
    for (int s = 0; s < NK; ++s) {
        Frag a1;
        a1.u[0] = *(const uint4*)(arow1 + s * 32);
        a1.u[1] = *(const uint4*)(arow1 + s * 32 + 16);
        acc1 = __builtin_amdgcn_wmma_f32_16x16x32_bf16(
            false, a1.v, false, bfr[s].v, (short)0, acc1, false, false);
    }

    // ---- epilogue: bias + store (C/D layout: row = i + (lane>=16)*8) ----
    const float bval  = bias ? bias[col0 + l16] : 0.f;
    const int   rbase = row0 + ((lane >> 4) << 3);
    #pragma unroll
    for (int i = 0; i < 8; ++i) {
        float v0 = acc0[i] + bval;
        float v1 = acc1[i] + bval;
        size_t off0 = (size_t)(rbase + i) * D_ + col0 + l16;
        size_t off1 = (size_t)(rbase + 16 + i) * D_ + col0 + l16;
        if (OUT_BF16) {
            ((bf16_t*)Cout)[off0] = f2bf(v0);
            ((bf16_t*)Cout)[off1] = f2bf(v1);
        } else {
            ((float*)Cout)[off0] = v0;
            ((float*)Cout)[off1] = v1;
        }
    }
}

// ---------------------------------------------------------------------------
// Weight packing: fp32 w[K x 128] (row-major) -> bf16 Wp[n*Kp + k] (col-major,
// zero padded to Kp) matching the B-fragment load pattern above.
// ---------------------------------------------------------------------------
__global__ void pack_w(const float* __restrict__ w, bf16_t* __restrict__ wp,
                       int K, int Kp)
{
    int idx = blockIdx.x * 256 + threadIdx.x;
    if (idx >= 128 * Kp) return;
    int n = idx / Kp, k = idx - n * Kp;
    float v = (k < K) ? w[(size_t)k * 128 + n] : 0.f;
    wp[idx] = f2bf(v);
}

// Build encoder input: [node(64) | global(16) | zeros(16)] as bf16, Kp=96
__global__ void prep_enc(const float* __restrict__ x, const float* __restrict__ g,
                         bf16_t* __restrict__ a)
{
    int idx = blockIdx.x * 256 + threadIdx.x;
    if (idx >= N_ * 96) return;
    int n = idx / 96, j = idx - n * 96;
    float v = (j < 64) ? x[(size_t)n * 64 + j] : ((j < 80) ? g[j - 64] : 0.f);
    a[idx] = f2bf(v);
}

__global__ void f32_to_bf(const float* __restrict__ x, bf16_t* __restrict__ y,
                          size_t n)
{
    size_t i = (size_t)blockIdx.x * 256 + threadIdx.x;
    if (i < n) y[i] = f2bf(x[i]);
}

__global__ void fill_u32k(unsigned* __restrict__ p, unsigned v, size_t n)
{
    size_t i = (size_t)blockIdx.x * 256 + threadIdx.x;
    if (i < n) p[i] = v;
}

// ---------------------------------------------------------------------------
// Row layernorm (one wave per row; optional broadcast add, relu, dual output)
// ---------------------------------------------------------------------------
__global__ __launch_bounds__(256) void ln_rows(
    const float* __restrict__ x, const float* __restrict__ addb,
    const float* __restrict__ scale, const float* __restrict__ bias,
    int relu, float* __restrict__ out_f32, bf16_t* __restrict__ out_bf,
    int nrows)
{
    int wid = threadIdx.x >> 5, lane = threadIdx.x & 31;
    int row = blockIdx.x * 8 + wid;
    if (row >= nrows) return;
    size_t base = (size_t)row * D_;
    float v[4];
    #pragma unroll
    for (int t = 0; t < 4; ++t) {
        int c = lane + 32 * t;
        v[t] = x[base + c] + (addb ? addb[c] : 0.f);
    }
    float s = wave_sum(v[0] + v[1] + v[2] + v[3]);
    float mu = s * (1.f / 128.f);
    float q = 0.f;
    #pragma unroll
    for (int t = 0; t < 4; ++t) { float d = v[t] - mu; q += d * d; }
    q = wave_sum(q);
    float inv = rsqrtf(q * (1.f / 128.f) + 1e-6f);
    #pragma unroll
    for (int t = 0; t < 4; ++t) {
        int c = lane + 32 * t;
        float y = (v[t] - mu) * inv * scale[c] + bias[c];
        if (relu) y = fmaxf(y, 0.f);
        if (out_f32) out_f32[base + c] = y;
        if (out_bf)  out_bf[base + c]  = f2bf(y);
    }
}

// layernorm + relu on a single 128-vector
__global__ void vec_ln_relu(const float* __restrict__ x,
                            const float* __restrict__ scale,
                            const float* __restrict__ bias,
                            float* __restrict__ out)
{
    __shared__ float sr[128];
    int t = threadIdx.x;
    float v = x[t];
    sr[t] = v; __syncthreads();
    for (int s = 64; s > 0; s >>= 1) { if (t < s) sr[t] += sr[t + s]; __syncthreads(); }
    float mu = sr[0] * (1.f / 128.f); __syncthreads();
    float d = v - mu;
    sr[t] = d * d; __syncthreads();
    for (int s = 64; s > 0; s >>= 1) { if (t < s) sr[t] += sr[t + s]; __syncthreads(); }
    float inv = rsqrtf(sr[0] * (1.f / 128.f) + 1e-6f);
    out[t] = fmaxf(d * inv * scale[t] + bias[t], 0.f);
}

// y[128] = x[128] * W[128x128] + b (+ optional residual)
__global__ void vec_dense128(const float* __restrict__ x, const float* __restrict__ w,
                             const float* __restrict__ b, const float* __restrict__ res,
                             float* __restrict__ y)
{
    int n = threadIdx.x;
    float acc = b[n];
    for (int k = 0; k < 128; ++k) acc = fmaf(x[k], w[k * 128 + n], acc);
    if (res) acc += res[n];
    y[n] = acc;
}

// nf = relu(a + b), dual f32/bf16 output
__global__ void relu_add_store(const float* __restrict__ a, const float* __restrict__ b,
                               float* __restrict__ of, bf16_t* __restrict__ ob, size_t n)
{
    size_t i = (size_t)blockIdx.x * 256 + threadIdx.x;
    if (i >= n) return;
    float v = fmaxf(a[i] + b[i], 0.f);
    of[i] = v; ob[i] = f2bf(v);
}

// ---------------------------------------------------------------------------
// Global-token multi-head attention over all nodes
// ---------------------------------------------------------------------------
__global__ void attn_scores(const float* __restrict__ q, const float* __restrict__ kbuf,
                            float* __restrict__ sc)
{
    int idx = blockIdx.x * 256 + threadIdx.x;
    if (idx >= N_ * H_) return;
    int n = idx >> 3, h = idx & 7;
    const float* kp = kbuf + (size_t)n * D_ + h * HD_;
    const float* qp = q + h * HD_;
    float s = 0.f;
    #pragma unroll
    for (int d = 0; d < HD_; ++d) s = fmaf(qp[d], kp[d], s);
    sc[idx] = s * 0.25f;                        // 1/sqrt(16)
}

__global__ __launch_bounds__(256) void attn_reduce(const float* __restrict__ sc,
                                                   const float* __restrict__ vbuf,
                                                   float* __restrict__ o)
{
    __shared__ float sr[256];
    int h = blockIdx.x, t = threadIdx.x;
    float mx = -3.4e38f;
    for (int n = t; n < N_; n += 256) mx = fmaxf(mx, sc[n * H_ + h]);
    sr[t] = mx; __syncthreads();
    for (int s = 128; s > 0; s >>= 1) { if (t < s) sr[t] = fmaxf(sr[t], sr[t + s]); __syncthreads(); }
    float gmax = sr[0]; __syncthreads();
    float lsum = 0.f, ov[HD_];
    #pragma unroll
    for (int d = 0; d < HD_; ++d) ov[d] = 0.f;
    for (int n = t; n < N_; n += 256) {
        float e = __expf(sc[n * H_ + h] - gmax);
        lsum += e;
        const float* vp = vbuf + (size_t)n * D_ + h * HD_;
        #pragma unroll
        for (int d = 0; d < HD_; ++d) ov[d] = fmaf(e, vp[d], ov[d]);
    }
    sr[t] = lsum; __syncthreads();
    for (int s = 128; s > 0; s >>= 1) { if (t < s) sr[t] += sr[t + s]; __syncthreads(); }
    float gsum = sr[0]; __syncthreads();
    for (int d = 0; d < HD_; ++d) {
        sr[t] = ov[d]; __syncthreads();
        for (int s = 128; s > 0; s >>= 1) { if (t < s) sr[t] += sr[t + s]; __syncthreads(); }
        if (t == 0) o[h * HD_ + d] = sr[0] / gsum;
        __syncthreads();
    }
}

// ---------------------------------------------------------------------------
// GATv2 edge passes
// ---------------------------------------------------------------------------
__global__ void edge_logits_max(const int2* __restrict__ el,
                                const float* __restrict__ hs,
                                const float* __restrict__ hr,
                                const bf16_t* __restrict__ he,
                                const float* __restrict__ attn,
                                float* __restrict__ logits,
                                unsigned* __restrict__ mmax)
{
    int idx = blockIdx.x * 256 + threadIdx.x;
    if (idx >= E_ * H_) return;
    int e = idx >> 3, h = idx & 7;
    int s = el[e].x, r = el[e].y;
    const float*  ps = hs + (size_t)s * D_ + h * HD_;
    const float*  pr = hr + (size_t)r * D_ + h * HD_;
    const bf16_t* pe = he + (size_t)e * D_ + h * HD_;
    const float*  pa = attn + h * HD_;
    float acc = 0.f;
    #pragma unroll
    for (int d = 0; d < HD_; ++d) {
        float m = ps[d] + pr[d] + bf2f(pe[d]);
        float lr = (m > 0.f) ? m : 0.2f * m;    // leaky relu
        acc = fmaf(lr, pa[d], acc);
    }
    logits[idx] = acc;
    atomicMax(mmax + r * H_ + h, f2ord(acc));
}

__global__ void edge_expsum(const int2* __restrict__ el,
                            const float* __restrict__ logits,
                            const unsigned* __restrict__ mmax,
                            float* __restrict__ exb, float* __restrict__ den)
{
    int idx = blockIdx.x * 256 + threadIdx.x;
    if (idx >= E_ * H_) return;
    int e = idx >> 3, h = idx & 7;
    int r = el[e].y;
    float ex = __expf(logits[idx] - ord2f(mmax[r * H_ + h]));
    exb[idx] = ex;
    atomicAdd(den + r * H_ + h, ex);
}

__global__ void edge_scatter(const int2* __restrict__ el,
                             const float* __restrict__ hs,
                             const float* __restrict__ exb,
                             const float* __restrict__ den,
                             float* __restrict__ agg)
{
    int idx = blockIdx.x * 256 + threadIdx.x;
    if (idx >= E_ * D_) return;
    int e = idx >> 7, j = idx & 127;
    int h = j >> 4;
    int s = el[e].x, r = el[e].y;
    float alpha = exb[e * H_ + h] / fmaxf(den[r * H_ + h], 1e-9f);
    atomicAdd(agg + (size_t)r * D_ + j, hs[(size_t)s * D_ + j] * alpha);
}

// ---------------------------------------------------------------------------
// Host side
// ---------------------------------------------------------------------------
static inline dim3 grid1(size_t n) { return dim3((unsigned)((n + 255) / 256)); }

struct AttnP { const float *qw, *qb, *kw, *kb, *vw, *vb, *ow, *ob; };
struct LayerP {
    const float *skw, *skb, *wgw, *wgb, *lns, *lnb;
    const float *wsw, *wsb, *wrw, *wrb, *wew, *web, *attn;
};

static void run_mha(const float* gin, float* gout, const AttnP& A,
                    const bf16_t* wpk, const bf16_t* wpv, const bf16_t* kvsrc,
                    float* kbuf, float* vbuf, float* sc,
                    float* qv, float* ov, float* gtmp,
                    const float* lns, const float* lnb, hipStream_t stream)
{
    vec_dense128<<<1, 128, 0, stream>>>(gin, A.qw, A.qb, nullptr, qv);
    gemm_bf16_128<128, false><<<N_ / 32, 256, 0, stream>>>(kvsrc, wpk, A.kb, kbuf);
    gemm_bf16_128<128, false><<<N_ / 32, 256, 0, stream>>>(kvsrc, wpv, A.vb, vbuf);
    attn_scores<<<grid1((size_t)N_ * H_), 256, 0, stream>>>(qv, kbuf, sc);
    attn_reduce<<<H_, 256, 0, stream>>>(sc, vbuf, ov);
    vec_dense128<<<1, 128, 0, stream>>>(ov, A.ow, A.ob, gin, gtmp); // residual
    vec_ln_relu<<<1, 128, 0, stream>>>(gtmp, lns, lnb, gout);
}

extern "C" void kernel_launch(void* const* d_in, const int* in_sizes, int n_in,
                              void* d_out, int out_size, void* d_ws, size_t ws_size,
                              hipStream_t stream)
{
    // ---- inputs (setup_inputs dict order; params flattened depth-first) ----
    const float* node_features = (const float*)d_in[0];   // N x 64
    const float* edge_features = (const float*)d_in[1];   // E x 32
    const int*   edge_list     = (const int*)d_in[2];     // E x 2
    const float* global_feat   = (const float*)d_in[3];   // 16
    int pi = 4;
    auto P = [&]() -> const float* { return (const float*)d_in[pi++]; };

    const float* g_token = P();
    const float* enc1_w = P(); const float* enc1_b = P();
    const float* ln1_s = P();  const float* ln1_b = P();
    const float* enc2_w = P(); const float* enc2_b = P();
    const float* ln2_s = P();  const float* ln2_b = P();
    AttnP a1; a1.qw = P(); a1.qb = P(); a1.kw = P(); a1.kb = P();
              a1.vw = P(); a1.vb = P(); a1.ow = P(); a1.ob = P();
    const float* lng1_s = P(); const float* lng1_b = P();
    const float* lnf_s  = P(); const float* lnf_b  = P();
    AttnP a2; a2.qw = P(); a2.qb = P(); a2.kw = P(); a2.kb = P();
              a2.vw = P(); a2.vb = P(); a2.ow = P(); a2.ob = P();
    const float* lng2_s = P(); const float* lng2_b = P();
    LayerP L[3];
    for (int i = 0; i < 3; ++i) {
        L[i].skw = P(); L[i].skb = P();
        L[i].wgw = P(); L[i].wgb = P();
        L[i].lns = P(); L[i].lnb = P();
        L[i].wsw = P(); L[i].wsb = P();
        L[i].wrw = P(); L[i].wrb = P();
        L[i].wew = P(); L[i].web = P();
        L[i].attn = P();
    }
    if (pi > n_in) return;

    // ---- workspace carve-up (~192 MB) ----
    char* base = (char*)d_ws;
    size_t used = 0;
    auto alloc = [&](size_t bytes) -> void* {
        void* p = base + used;
        used += (bytes + 255) & ~(size_t)255;
        return p;
    };
    float*  nf    = (float*)alloc((size_t)N_ * D_ * 4);
    bf16_t* nfb   = (bf16_t*)alloc((size_t)N_ * D_ * 2);
    float*  skip  = (float*)alloc((size_t)N_ * D_ * 4);   // also encoder tmp
    float*  hs    = (float*)alloc((size_t)N_ * D_ * 4);   // also attn K
    float*  hr    = (float*)alloc((size_t)N_ * D_ * 4);   // also attn V
    float*  agg   = (float*)alloc((size_t)N_ * D_ * 4);
    bf16_t* xb    = (bf16_t*)alloc((size_t)N_ * D_ * 2);  // x1 / nf_ln / nf_final
    bf16_t* aenc  = (bf16_t*)alloc((size_t)N_ * 96 * 2);
    bf16_t* efb   = (bf16_t*)alloc((size_t)E_ * 32 * 2);
    bf16_t* heb   = (bf16_t*)alloc((size_t)E_ * D_ * 2);
    float*  logits= (float*)alloc((size_t)E_ * H_ * 4);
    float*  exb   = (float*)alloc((size_t)E_ * H_ * 4);
    unsigned* mmax= (unsigned*)alloc((size_t)N_ * H_ * 4);
    float*  den   = (float*)alloc((size_t)N_ * H_ * 4);
    float*  sc    = (float*)alloc((size_t)N_ * H_ * 4);
    float*  qv    = (float*)alloc(128 * 4);
    float*  ov    = (float*)alloc(128 * 4);
    float*  gcur  = (float*)alloc(128 * 4);
    float*  gvec  = (float*)alloc(128 * 4);
    float*  gtmp  = (float*)alloc(128 * 4);
    bf16_t* wp_enc1 = (bf16_t*)alloc(96 * 128 * 2);
    bf16_t* wp_enc2 = (bf16_t*)alloc(128 * 128 * 2);
    bf16_t* wp_a1k  = (bf16_t*)alloc(128 * 128 * 2);
    bf16_t* wp_a1v  = (bf16_t*)alloc(128 * 128 * 2);
    bf16_t* wp_a2k  = (bf16_t*)alloc(128 * 128 * 2);
    bf16_t* wp_a2v  = (bf16_t*)alloc(128 * 128 * 2);
    bf16_t *wp_sk[3], *wp_ws[3], *wp_wr[3], *wp_we[3];
    for (int i = 0; i < 3; ++i) {
        wp_sk[i] = (bf16_t*)alloc(128 * 128 * 2);
        wp_ws[i] = (bf16_t*)alloc(128 * 128 * 2);
        wp_wr[i] = (bf16_t*)alloc(128 * 128 * 2);
        wp_we[i] = (bf16_t*)alloc(32 * 128 * 2);
    }
    if (used > ws_size) return;   // workspace too small: avoid OOB

    const int2* el = (const int2*)edge_list;

    // ---- pack weights / prep activations ----
    pack_w<<<grid1(128 * 96), 256, 0, stream>>>(enc1_w, wp_enc1, 80, 96);
    pack_w<<<grid1(128 * 128), 256, 0, stream>>>(enc2_w, wp_enc2, 128, 128);
    pack_w<<<grid1(128 * 128), 256, 0, stream>>>(a1.kw, wp_a1k, 128, 128);
    pack_w<<<grid1(128 * 128), 256, 0, stream>>>(a1.vw, wp_a1v, 128, 128);
    pack_w<<<grid1(128 * 128), 256, 0, stream>>>(a2.kw, wp_a2k, 128, 128);
    pack_w<<<grid1(128 * 128), 256, 0, stream>>>(a2.vw, wp_a2v, 128, 128);
    for (int i = 0; i < 3; ++i) {
        pack_w<<<grid1(128 * 128), 256, 0, stream>>>(L[i].skw, wp_sk[i], 128, 128);
        pack_w<<<grid1(128 * 128), 256, 0, stream>>>(L[i].wsw, wp_ws[i], 128, 128);
        pack_w<<<grid1(128 * 128), 256, 0, stream>>>(L[i].wrw, wp_wr[i], 128, 128);
        pack_w<<<grid1(128 * 32), 256, 0, stream>>>(L[i].wew, wp_we[i], 32, 32);
    }
    prep_enc<<<grid1((size_t)N_ * 96), 256, 0, stream>>>(node_features, global_feat, aenc);
    f32_to_bf<<<grid1((size_t)E_ * 32), 256, 0, stream>>>(edge_features, efb, (size_t)E_ * 32);

    // ---- encoder ----
    gemm_bf16_128<96, false><<<N_ / 32, 256, 0, stream>>>(aenc, wp_enc1, enc1_b, nf);
    ln_rows<<<(N_ + 7) / 8, 256, 0, stream>>>(nf, nullptr, ln1_s, ln1_b, 1, nullptr, xb, N_);
    gemm_bf16_128<128, false><<<N_ / 32, 256, 0, stream>>>(xb, wp_enc2, enc2_b, skip);
    ln_rows<<<(N_ + 7) / 8, 256, 0, stream>>>(skip, nullptr, ln2_s, ln2_b, 0, nf, nfb, N_);

    // ---- attention block 1 (global token over nodes) ----
    run_mha(g_token, gcur, a1, wp_a1k, wp_a1v, nfb, hs, hr, sc, qv, ov, gtmp,
            lng1_s, lng1_b, stream);

    // ---- GAT layers ----
    for (int i = 0; i < 3; ++i) {
        gemm_bf16_128<128, false><<<N_ / 32, 256, 0, stream>>>(nfb, wp_sk[i], L[i].skb, skip);
        vec_dense128<<<1, 128, 0, stream>>>(gcur, L[i].wgw, L[i].wgb, nullptr, gvec);
        ln_rows<<<(N_ + 7) / 8, 256, 0, stream>>>(nf, gvec, L[i].lns, L[i].lnb, 0,
                                                  nullptr, xb, N_);
        gemm_bf16_128<128, false><<<N_ / 32, 256, 0, stream>>>(xb, wp_ws[i], L[i].wsb, hs);
        gemm_bf16_128<128, false><<<N_ / 32, 256, 0, stream>>>(xb, wp_wr[i], L[i].wrb, hr);
        gemm_bf16_128<32, true><<<E_ / 32, 256, 0, stream>>>(efb, wp_we[i], L[i].web, heb);

        fill_u32k<<<grid1((size_t)N_ * H_), 256, 0, stream>>>(mmax, 0x007FFFFFu, (size_t)N_ * H_);
        fill_u32k<<<grid1((size_t)N_ * H_), 256, 0, stream>>>((unsigned*)den, 0u, (size_t)N_ * H_);
        fill_u32k<<<grid1((size_t)N_ * D_), 256, 0, stream>>>((unsigned*)agg, 0u, (size_t)N_ * D_);

        edge_logits_max<<<grid1((size_t)E_ * H_), 256, 0, stream>>>(el, hs, hr, heb,
                                                                    L[i].attn, logits, mmax);
        edge_expsum<<<grid1((size_t)E_ * H_), 256, 0, stream>>>(el, logits, mmax, exb, den);
        edge_scatter<<<grid1((size_t)E_ * D_), 256, 0, stream>>>(el, hs, exb, den, agg);

        relu_add_store<<<grid1((size_t)N_ * D_), 256, 0, stream>>>(agg, skip, nf, nfb,
                                                                   (size_t)N_ * D_);
    }

    // ---- final LN + attention block 2 -> output ----
    ln_rows<<<(N_ + 7) / 8, 256, 0, stream>>>(nf, nullptr, lnf_s, lnf_b, 0, nullptr, xb, N_);
    run_mha(gcur, (float*)d_out, a2, wp_a2k, wp_a2v, xb, hs, hr, sc, qv, ov, gtmp,
            lng2_s, lng2_b, stream);
}
